// MultiQueryAttention_20229295964202
// MI455X (gfx1250) — matrix operand order repaired and checked by reading, not compiled, not run on hardware
//
#include <hip/hip_runtime.h>
#include <hip/hip_bf16.h>

// Problem constants (from reference)
#define B_   4
#define SQ_  2048
#define SK_  2048
#define H_   8
#define D_   512
#define P_   64

typedef __attribute__((ext_vector_type(16))) _Float16     v16h;
typedef __attribute__((ext_vector_type(8)))  float        v8f;
typedef __attribute__((ext_vector_type(2)))  _Float16     h2;
typedef __attribute__((ext_vector_type(4)))  unsigned int u32x4;   // POD 16B move
typedef __attribute__((ext_vector_type(4)))  int          i32x4;

// Async global->LDS copy (CDNA5 GLOBAL_LOAD_ASYNC_TO_LDS_B128, ASYNCcnt),
// guarded so the kernel still compiles if the toolchain lacks the builtins.
#if defined(__has_builtin)
#  if __has_builtin(__builtin_amdgcn_global_load_async_to_lds_b128) && \
      __has_builtin(__builtin_amdgcn_s_wait_asynccnt)
#    define USE_ASYNC_COPY 1
#  endif
#endif
#ifndef USE_ASYNC_COPY
#  define USE_ASYNC_COPY 0
#endif

#if USE_ASYNC_COPY
// Builtin signature (from hipcc diagnostic): (V4i AS1*, V4i AS3*, Imm, Imm)
typedef __attribute__((address_space(1))) i32x4* gptr_b128;
typedef __attribute__((address_space(3))) i32x4* lptr_b128;
__device__ __forceinline__ void async_copy16(const _Float16* g, _Float16* l) {
  __builtin_amdgcn_global_load_async_to_lds_b128((gptr_b128)g, (lptr_b128)l, 0, 0);
}
#endif

// ---------------------------------------------------------------------------
// WMMA helpers (CDNA5 wave32, V_WMMA_F32_16X16X32_F16)
// ---------------------------------------------------------------------------
__device__ __forceinline__ v8f wmma16(v16h a, v16h b, v8f c) {
  return __builtin_amdgcn_wmma_f32_16x16x32_f16(false, a, false, b, (short)0, c,
                                                false, false);
}

// A-fragment (16x32 f16, row-major, leading dim lda): 2x contiguous 16B reads.
__device__ __forceinline__ v16h load_a_frag(const _Float16* A, int lda, int lane) {
  union { u32x4 q[2]; v16h v; } u;
  const _Float16* Ar = A + (lane & 15) * lda + ((lane >> 4) << 3);
  u.q[0] = *(const u32x4*)(Ar);
  u.q[1] = *(const u32x4*)(Ar + 16);
  return u.v;
}

// B-fragment (32x16 f16) from transposed source: B[k][n] = Bt[n*ldb + k].
__device__ __forceinline__ v16h load_bt_frag(const _Float16* Bt, int ldb, int lane) {
  union { u32x4 q[2]; v16h v; } u;
  const _Float16* Bp = Bt + (lane & 15) * ldb + ((lane >> 4) << 3);
  u.q[0] = *(const u32x4*)(Bp);
  u.q[1] = *(const u32x4*)(Bp + 16);
  return u.v;
}

// ---------------------------------------------------------------------------
// Kernel 0: one-time weight prep. All W -> f16, transposed [n(64)][k(512)].
// Slices: 0..7 = Wq head h, 8 = Wk, 9 = Wv, 10 = Wo.
// ---------------------------------------------------------------------------
__global__ __launch_bounds__(256) void prep_weights_kernel(
    const float* Wq, const float* Wk, const float* Wv, const float* Wo,
    _Float16* WqT, _Float16* WkT, _Float16* WvT, _Float16* WoT) {
  const int slice = blockIdx.y;
  const float* src;
  _Float16* dst;
  if (slice < 8)       { src = Wq + (size_t)slice * D_ * 64; dst = WqT + (size_t)slice * 64 * D_; }
  else if (slice == 8) { src = Wk; dst = WkT; }
  else if (slice == 9) { src = Wv; dst = WvT; }
  else                 { src = Wo; dst = WoT; }
  const int i = blockIdx.x * 256 + threadIdx.x;   // 0 .. 64*512-1
  const int n = i >> 9, k = i & 511;
  dst[(size_t)n * D_ + k] = (_Float16)src[(size_t)k * 64 + n];
}

// ---------------------------------------------------------------------------
// Block GEMM core: each wave accumulates a 16x64 tile of X(16x512)@W(512x64).
// Wt is f16 transposed [64][512]. 4 waves share the Wt chunk in LDS.
// Register double-buffering: next chunk's global loads overlap the WMMAs.
// ---------------------------------------------------------------------------
template <typename TIN>
__device__ __forceinline__ void gemm16x64_acc(const TIN* Xr,          // 16 x D_, stride D_
                                              const _Float16* Wt,     // [64][512]
                                              _Float16* Alds,         // per-wave 16*32
                                              _Float16* Btlds,        // shared 64*32
                                              v8f acc[4]) {
  const int tid = threadIdx.x, lane = tid & 31;
  const int arow = lane >> 1, ach = (lane & 1) << 4;     // each lane: half a row
  const TIN* Ap = Xr + (size_t)arow * D_ + ach;

  u32x4 bq0, bq1;            // staged Wt chunk (2 x 16B per thread)
  u32x4 ah0, ah1;            // staged A (f16 path)
  float af[16];              // staged A (f32 path)

  auto fetch = [&](int kc) {
    bq0 = *(const u32x4*)(Wt + (size_t)(tid >> 2) * D_ + kc + (tid & 3) * 8);
    const int i1 = tid + 128;
    bq1 = *(const u32x4*)(Wt + (size_t)(i1 >> 2) * D_ + kc + (i1 & 3) * 8);
    if constexpr (sizeof(TIN) == 2) {
      ah0 = *(const u32x4*)(Ap + kc);
      ah1 = *(const u32x4*)(Ap + kc + 8);
    } else {
#pragma unroll
      for (int j = 0; j < 16; ++j) af[j] = (float)Ap[kc + j];
    }
  };

  fetch(0);
  for (int kc = 0; kc < D_; kc += 32) {
    __syncthreads();
    ((u32x4*)Btlds)[tid]       = bq0;
    ((u32x4*)Btlds)[tid + 128] = bq1;
    {
      u32x4* ad = (u32x4*)(Alds + arow * 32 + ach);
      if constexpr (sizeof(TIN) == 2) {
        ad[0] = ah0; ad[1] = ah1;
      } else {
        union { _Float16 h[16]; u32x4 q[2]; } ua;
#pragma unroll
        for (int j = 0; j < 16; ++j) ua.h[j] = (_Float16)af[j];
        ad[0] = ua.q[0]; ad[1] = ua.q[1];
      }
    }
    __syncthreads();
    if (kc + 32 < D_) fetch(kc + 32);   // overlap next chunk with compute
    v16h a = load_a_frag(Alds, 32, lane);
    acc[0] = wmma16(a, load_bt_frag(Btlds + 0 * 16 * 32, 32, lane), acc[0]);
    acc[1] = wmma16(a, load_bt_frag(Btlds + 1 * 16 * 32, 32, lane), acc[1]);
    acc[2] = wmma16(a, load_bt_frag(Btlds + 2 * 16 * 32, 32, lane), acc[2]);
    acc[3] = wmma16(a, load_bt_frag(Btlds + 3 * 16 * 32, 32, lane), acc[3]);
  }
}

// Key permutation within a 32-key block: key = col + 16*half -> 2*col + half.
// Applied to P writes (enables packed 32-bit stores) and to Vt's key axis,
// leaving the P@V contraction invariant.
__device__ __forceinline__ size_t perm_key(size_t key) {
  const size_t j = key & 31;
  return (key & ~(size_t)31) | (((j & 15) << 1) | (j >> 4));
}

// ---------------------------------------------------------------------------
// Kernel 1: shared K / V projections. K -> row-major Kh[B][SK][64] (f16).
// V -> transposed Vt[B][64][SK] (f16) with perm_key applied to the key axis.
// ---------------------------------------------------------------------------
__global__ __launch_bounds__(128) void proj_kv_kernel(const float* store_seq,
                                                      const _Float16* WkT, const float* bk,
                                                      const _Float16* WvT, const float* bv,
                                                      _Float16* Kh, _Float16* Vt) {
  __shared__ __align__(16) _Float16 Btlds[64 * 32];
  __shared__ __align__(16) _Float16 Alds[4][16 * 32];
  const int wave = threadIdx.x >> 5, lane = threadIdx.x & 31;
  const size_t row0 = ((size_t)blockIdx.x * 4 + wave) * 16;   // flattened [B*SK]
  const bool isV = blockIdx.y != 0;

  v8f acc[4] = {};
  gemm16x64_acc<float>(store_seq + row0 * D_, isV ? WvT : WkT, Alds[wave], Btlds, acc);

  const float* bb = isV ? bv : bk;
  const int col = lane & 15, rowoff = (lane >> 4) << 3;
  const float b0 = bb[col], b1 = bb[16 + col], b2 = bb[32 + col], b3 = bb[48 + col];
  if (!isV) {
    _Float16* Oh = Kh + row0 * 64;
#pragma unroll
    for (int r = 0; r < 8; ++r) {
      _Float16* Or = Oh + (size_t)(r + rowoff) * 64;
      Or[col]      = (_Float16)(acc[0][r] + b0);
      Or[16 + col] = (_Float16)(acc[1][r] + b1);
      Or[32 + col] = (_Float16)(acc[2][r] + b2);
      Or[48 + col] = (_Float16)(acc[3][r] + b3);
    }
  } else {
    const int b = (int)(row0 / SK_);
    const size_t s = row0 % SK_;
    _Float16* VtB = Vt + (size_t)b * 64 * SK_;
#pragma unroll
    for (int r = 0; r < 8; ++r) {
      const size_t srow = perm_key(s + r + rowoff);
      VtB[(size_t)(col)      * SK_ + srow] = (_Float16)(acc[0][r] + b0);
      VtB[(size_t)(16 + col) * SK_ + srow] = (_Float16)(acc[1][r] + b1);
      VtB[(size_t)(32 + col) * SK_ + srow] = (_Float16)(acc[2][r] + b2);
      VtB[(size_t)(48 + col) * SK_ + srow] = (_Float16)(acc[3][r] + b3);
    }
  }
}

// ---------------------------------------------------------------------------
// Kernel 2: per-head Q projection; folds the 1/sqrt(P) score scale into Q.
// ---------------------------------------------------------------------------
__global__ __launch_bounds__(128) void proj_q_kernel(const float* query_seq,
                                                     const _Float16* WqT, const float* bq,
                                                     _Float16* Qh) {
  __shared__ __align__(16) _Float16 Btlds[64 * 32];
  __shared__ __align__(16) _Float16 Alds[4][16 * 32];
  const int wave = threadIdx.x >> 5, lane = threadIdx.x & 31;
  const int h = blockIdx.y;
  const size_t row0 = ((size_t)blockIdx.x * 4 + wave) * 16;   // flattened [B*SQ]
  const int b = (int)(row0 / SQ_);
  const size_t s = row0 % SQ_;

  v8f acc[4] = {};
  gemm16x64_acc<float>(query_seq + row0 * D_, WqT + (size_t)h * 64 * D_,
                       Alds[wave], Btlds, acc);

  const float* bb = bq + (size_t)h * 64;
  const int col = lane & 15, rowoff = (lane >> 4) << 3;
  const float b0 = bb[col], b1 = bb[16 + col], b2 = bb[32 + col], b3 = bb[48 + col];
  const float sc = 0.125f;   // 1/sqrt(64), folded here
  _Float16* Oh = Qh + (((size_t)b * H_ + h) * SQ_ + s) * 64;
#pragma unroll
  for (int r = 0; r < 8; ++r) {
    _Float16* Or = Oh + (size_t)(r + rowoff) * 64;
    Or[col]      = (_Float16)((acc[0][r] + b0) * sc);
    Or[16 + col] = (_Float16)((acc[1][r] + b1) * sc);
    Or[32 + col] = (_Float16)((acc[2][r] + b2) * sc);
    Or[48 + col] = (_Float16)((acc[3][r] + b3) * sc);
  }
}

// ---------------------------------------------------------------------------
// Kernel 3: flash attention. 4 waves x 16 query rows share each 32-key tile.
// K/V tiles double-buffered in LDS via the CDNA5 async copy engine (ASYNCcnt)
// when available; register-staged fallback otherwise.
// ---------------------------------------------------------------------------
__global__ __launch_bounds__(128) void attn_kernel(const _Float16* Qh,
                                                   const _Float16* Kh,
                                                   const _Float16* Vt,
                                                   _Float16* Ch) {
  __shared__ __align__(16) _Float16 Klds[2][32 * 64];    // [key][p]
  __shared__ __align__(16) _Float16 Vtlds[2][64 * 32];   // [p][key'] (permuted)
  __shared__ __align__(16) _Float16 Qlds[4][16 * 64];
  __shared__ __align__(16) _Float16 Plds[4][16 * 32];

  const int tid = threadIdx.x, wave = tid >> 5, lane = tid & 31;
  const int nQB = SQ_ / 64;
  const int qb = blockIdx.x % nQB;
  const int bh = blockIdx.x / nQB;
  const int h = bh % H_, b = bh / H_;

  const size_t qrow0 = (size_t)qb * 64 + (size_t)wave * 16;
  const _Float16* Qg = Qh + (((size_t)b * H_ + h) * SQ_ + qrow0) * 64;
  {
    const u32x4* src = (const u32x4*)Qg;
    u32x4* dst = (u32x4*)Qlds[wave];
    for (int i = lane; i < 128; i += 32) dst[i] = src[i];
  }
  __syncthreads();
  const v16h aq0 = load_a_frag(Qlds[wave], 64, lane);        // p 0..31
  const v16h aq1 = load_a_frag(Qlds[wave] + 32, 64, lane);   // p 32..63

  float m[8], l[8];
  v8f o0 = {}, o1 = {}, o2 = {}, o3 = {};
#pragma unroll
  for (int r = 0; r < 8; ++r) { m[r] = -1e30f; l[r] = 0.0f; }

  const _Float16* Kg  = Kh + (size_t)b * SK_ * 64;
  const _Float16* Vtg = Vt + (size_t)b * 64 * SK_;
  const int i1 = tid + 128;
  const int NT = SK_ / 32;

#if USE_ASYNC_COPY
  auto issue_tile = [&](int kt, int buf) {   // 4 async b128 ops per wave
    const _Float16* ksrc = Kg + (size_t)kt * 2048;
    const _Float16* vsrc = Vtg + (size_t)kt * 32;
    async_copy16(ksrc + (size_t)tid * 8, Klds[buf] + tid * 8);
    async_copy16(ksrc + (size_t)i1 * 8,  Klds[buf] + i1 * 8);
    async_copy16(vsrc + (size_t)(tid >> 2) * SK_ + (tid & 3) * 8, Vtlds[buf] + tid * 8);
    async_copy16(vsrc + (size_t)(i1 >> 2) * SK_ + (i1 & 3) * 8,  Vtlds[buf] + i1 * 8);
  };
  issue_tile(0, 0);
#else
  u32x4 kr0, kr1, vr0, vr1;
  auto fetch = [&](int kt) {
    const u32x4* ks = (const u32x4*)(Kg + (size_t)kt * 2048);
    kr0 = ks[tid];
    kr1 = ks[i1];
    vr0 = *(const u32x4*)(Vtg + (size_t)(tid >> 2) * SK_ + (size_t)kt * 32 + (tid & 3) * 8);
    vr1 = *(const u32x4*)(Vtg + (size_t)(i1 >> 2) * SK_ + (size_t)kt * 32 + (i1 & 3) * 8);
  };
  fetch(0);
#endif

  for (int kt = 0; kt < NT; ++kt) {
#if USE_ASYNC_COPY
    const int cur = kt & 1;
    __syncthreads();                               // ping buffer reads retired
    const int nxt = (kt + 1 < NT) ? (kt + 1) : kt; // dummy reissue on last iter
    issue_tile(nxt, cur ^ 1);                      // keeps ASYNCcnt target == 4
    __builtin_amdgcn_s_wait_asynccnt(4);           // own tile-kt portion done
    __syncthreads();                               // everyone's portion visible
    const _Float16* Kcur = Klds[cur];
    const _Float16* Vcur = Vtlds[cur];
    if (kt + 2 < NT)
      __builtin_prefetch(Kg + (size_t)(kt + 2) * 2048 + (size_t)tid * 16, 0, 1);
#else
    __syncthreads();                       // prior tile's LDS reads retired
    ((u32x4*)Klds[0])[tid]  = kr0;
    ((u32x4*)Klds[0])[i1]   = kr1;
    ((u32x4*)Vtlds[0])[tid] = vr0;
    ((u32x4*)Vtlds[0])[i1]  = vr1;
    __syncthreads();
    if (kt + 1 < NT) {
      fetch(kt + 1);                       // overlap with compute below
      if (kt + 2 < NT)
        __builtin_prefetch(Kg + (size_t)(kt + 2) * 2048 + (size_t)tid * 16, 0, 1);
    }
    const _Float16* Kcur = Klds[0];
    const _Float16* Vcur = Vtlds[0];
#endif

    // S = Q K^T : 16 rows x 32 keys (scale pre-folded into Q)
    v8f s0 = {}, s1 = {};
    s0 = wmma16(aq0, load_bt_frag(Kcur, 64, lane), s0);
    s0 = wmma16(aq1, load_bt_frag(Kcur + 32, 64, lane), s0);
    s1 = wmma16(aq0, load_bt_frag(Kcur + 16 * 64, 64, lane), s1);
    s1 = wmma16(aq1, load_bt_frag(Kcur + 16 * 64 + 32, 64, lane), s1);

    // Online softmax; half-wave shuffle reductions (xor 1,2,4,8)
#pragma unroll
    for (int r = 0; r < 8; ++r) {
      const float x0 = s0[r], x1 = s1[r];
      float mx = fmaxf(x0, x1);
      mx = fmaxf(mx, __shfl_xor(mx, 1));
      mx = fmaxf(mx, __shfl_xor(mx, 2));
      mx = fmaxf(mx, __shfl_xor(mx, 4));
      mx = fmaxf(mx, __shfl_xor(mx, 8));
      const float mn = fmaxf(m[r], mx);
      const float alpha = __expf(m[r] - mn);
      const float p0 = __expf(x0 - mn);
      const float p1 = __expf(x1 - mn);
      float rs = p0 + p1;
      rs += __shfl_xor(rs, 1);
      rs += __shfl_xor(rs, 2);
      rs += __shfl_xor(rs, 4);
      rs += __shfl_xor(rs, 8);
      m[r] = mn;
      l[r] = l[r] * alpha + rs;
      o0[r] *= alpha; o1[r] *= alpha; o2[r] *= alpha; o3[r] *= alpha;
      s0[r] = p0; s1[r] = p1;
    }

    // C->A transpose of P through per-wave LDS; packed 32-bit stores at
    // permuted key index 2*col+half (matches Vt's permuted key axis).
    {
      const int col = lane & 15, rowoff = (lane >> 4) << 3;
      _Float16* P = Plds[wave];
#pragma unroll
      for (int r = 0; r < 8; ++r) {
        h2 pk; pk[0] = (_Float16)s0[r]; pk[1] = (_Float16)s1[r];
        *(h2*)(P + (r + rowoff) * 32 + 2 * col) = pk;
      }
    }
    const v16h ap = load_a_frag(Plds[wave], 32, lane);   // same-wave DS order

    // O += P @ V  (Vcur rows are [p][key'], contiguous b128 fragments)
    o0 = wmma16(ap, load_bt_frag(Vcur + 0 * 16 * 32, 32, lane), o0);
    o1 = wmma16(ap, load_bt_frag(Vcur + 1 * 16 * 32, 32, lane), o1);
    o2 = wmma16(ap, load_bt_frag(Vcur + 2 * 16 * 32, 32, lane), o2);
    o3 = wmma16(ap, load_bt_frag(Vcur + 3 * 16 * 32, 32, lane), o3);
  }

  // Normalize, write head-concat buffer Ch[B][SQ][H*P]
  _Float16* Cg = Ch + ((size_t)b * SQ_ + qrow0) * (H_ * P_) + (size_t)h * P_;
  const int col = lane & 15, rowoff = (lane >> 4) << 3;
#pragma unroll
  for (int r = 0; r < 8; ++r) {
    const float inv = 1.0f / l[r];
    _Float16* Or = Cg + (size_t)(r + rowoff) * (H_ * P_);
    Or[col]      = (_Float16)(o0[r] * inv);
    Or[16 + col] = (_Float16)(o1[r] * inv);
    Or[32 + col] = (_Float16)(o2[r] * inv);
    Or[48 + col] = (_Float16)(o3[r] * inv);
  }
}

// ---------------------------------------------------------------------------
// Kernel 4: output projection concat(f16) @ Wo + bo -> f32
// ---------------------------------------------------------------------------
__global__ __launch_bounds__(128) void out_proj_kernel(const _Float16* Ch,
                                                       const _Float16* WoT,
                                                       const float* bo,
                                                       float* out) {
  __shared__ __align__(16) _Float16 Btlds[64 * 32];
  __shared__ __align__(16) _Float16 Alds[4][16 * 32];
  const int wave = threadIdx.x >> 5, lane = threadIdx.x & 31;
  const size_t row0 = ((size_t)blockIdx.x * 4 + wave) * 16;   // flattened [B*SQ]

  v8f acc[4] = {};
  gemm16x64_acc<_Float16>(Ch + row0 * D_, WoT, Alds[wave], Btlds, acc);

  const int col = lane & 15, rowoff = (lane >> 4) << 3;
  const float b0 = bo[col], b1 = bo[16 + col], b2 = bo[32 + col], b3 = bo[48 + col];
  float* Og = out + row0 * 64;
#pragma unroll
  for (int r = 0; r < 8; ++r) {
    float* Or = Og + (size_t)(r + rowoff) * 64;
    Or[col]      = acc[0][r] + b0;
    Or[16 + col] = acc[1][r] + b1;
    Or[32 + col] = acc[2][r] + b2;
    Or[48 + col] = acc[3][r] + b3;
  }
}

// ---------------------------------------------------------------------------
// Launch
// ---------------------------------------------------------------------------
extern "C" void kernel_launch(void* const* d_in, const int* in_sizes, int n_in,
                              void* d_out, int out_size, void* d_ws, size_t ws_size,
                              hipStream_t stream) {
  const float* query = (const float*)d_in[0];
  const float* store = (const float*)d_in[1];
  const float* Wq = (const float*)d_in[2];
  const float* bq = (const float*)d_in[3];
  const float* Wk = (const float*)d_in[4];
  const float* bk = (const float*)d_in[5];
  const float* Wv = (const float*)d_in[6];
  const float* bv = (const float*)d_in[7];
  const float* Wo = (const float*)d_in[8];
  const float* bo = (const float*)d_in[9];
  float* out = (float*)d_out;

  char* ws = (char*)d_ws;
  _Float16* Qh  = (_Float16*)ws; ws += (size_t)B_ * H_ * SQ_ * P_ * 2;  // 8 MB
  _Float16* Kh  = (_Float16*)ws; ws += (size_t)B_ * SK_ * P_ * 2;       // 1 MB
  _Float16* Vt  = (_Float16*)ws; ws += (size_t)B_ * P_ * SK_ * 2;       // 1 MB
  _Float16* Ch  = (_Float16*)ws; ws += (size_t)B_ * SQ_ * D_ * 2;       // 8 MB
  _Float16* WqT = (_Float16*)ws; ws += (size_t)H_ * 64 * D_ * 2;        // 512 KB
  _Float16* WkT = (_Float16*)ws; ws += (size_t)64 * D_ * 2;             // 64 KB
  _Float16* WvT = (_Float16*)ws; ws += (size_t)64 * D_ * 2;             // 64 KB
  _Float16* WoT = (_Float16*)ws;                                        // 64 KB

  prep_weights_kernel<<<dim3(128, 11), 256, 0, stream>>>(Wq, Wk, Wv, Wo,
                                                         WqT, WkT, WvT, WoT);
  proj_kv_kernel<<<dim3(B_ * SK_ / 64, 2), 128, 0, stream>>>(store, WkT, bk, WvT, bv, Kh, Vt);
  proj_q_kernel<<<dim3(B_ * SQ_ / 64, H_), 128, 0, stream>>>(query, WqT, bq, Qh);
  attn_kernel<<<dim3(B_ * H_ * (SQ_ / 64)), 128, 0, stream>>>(Qh, Kh, Vt, Ch);
  out_proj_kernel<<<dim3(B_ * SQ_ / 64), 128, 0, stream>>>(Ch, WoT, bo, out);
}
